// GAT3_38465727103215
// MI455X (gfx1250) — matrix operand order repaired and checked by reading, not compiled, not run on hardware
//
#include <hip/hip_runtime.h>
#include <hip/hip_bf16.h>
#include <stdint.h>
#include <stddef.h>

#define N_NODES 50000
#define N_EDGES 800000
#define FIN     256
#define HEADS   4
#define DHEAD   128
#define FOUT    (HEADS * DHEAD)   // 512

typedef __attribute__((ext_vector_type(16))) __bf16         v16bf;
typedef __attribute__((ext_vector_type(8)))  float          v8f;
typedef __attribute__((ext_vector_type(8)))  unsigned short u16x8;
typedef __attribute__((ext_vector_type(16))) unsigned short u16x16;

// ---------- helpers ----------
__device__ __forceinline__ unsigned short f2bf(float f) {
  unsigned int u = __float_as_uint(f);
  u += 0x7FFFu + ((u >> 16) & 1u);       // round-to-nearest-even
  return (unsigned short)(u >> 16);
}
// monotone float <-> uint mapping so segment-max can use global_atomic_max_u32
__device__ __forceinline__ unsigned int ford(float f) {
  unsigned int u = __float_as_uint(f);
  return (u & 0x80000000u) ? ~u : (u | 0x80000000u);
}
__device__ __forceinline__ float funord(unsigned int u) {
  unsigned int v = (u & 0x80000000u) ? (u & 0x7FFFFFFFu) : ~u;
  return __uint_as_float(v);
}

// ---------- stage 0: f32 -> bf16 convert ----------
__global__ void k_f32_to_bf16(const float* __restrict__ in,
                              unsigned short* __restrict__ out, long long n) {
  long long i = (long long)blockIdx.x * blockDim.x + threadIdx.x;
  if (i < n) out[i] = f2bf(in[i]);
}

// ---------- stage 0b: pack W [K, FOUT] into per-lane WMMA B layout ----------
// B tile (32x16 bf16): lanes 0-15 -> K rows kt*32+{0..7,16..23}, lanes 16-31 -> +8; N = lane&15.
// Storage: Bpk[((nt*Ktiles + kt)*32 + lane)*16 + i]
__global__ void k_pack_b(const float* __restrict__ W, unsigned short* __restrict__ Bpk,
                         int Ktiles, int Ntiles, int Nout) {
  int t = blockIdx.x * blockDim.x + threadIdx.x;
  int total = Ntiles * Ktiles * 32;
  if (t >= total) return;
  int lane = t & 31;
  int rest = t >> 5;
  int kt = rest % Ktiles;
  int nt = rest / Ktiles;
  int kb  = (lane & 16) ? 8 : 0;
  int col = nt * 16 + (lane & 15);
  unsigned short* o = Bpk + (size_t)t * 16;
#pragma unroll
  for (int i = 0; i < 16; ++i) {
    int krow = kt * 32 + kb + i + ((i >= 8) ? 8 : 0);
    o[i] = f2bf(W[(size_t)krow * Nout + col]);
  }
}

// ---------- stage 1: WMMA GEMM  C[M,Nout] = A(bf16)[M,K] * B(bf16 packed) ----------
__global__ void __launch_bounds__(256)
k_gemm_wmma_bf16(const unsigned short* __restrict__ A,
                 const unsigned short* __restrict__ Bpk,
                 float* __restrict__ C,
                 int Mtiles, int Ktiles, int Ntiles, int K, int Nout) {
  const int lane = threadIdx.x & 31;
  const int wave = blockIdx.x * (blockDim.x >> 5) + (threadIdx.x >> 5);
  if (wave >= Mtiles * Ntiles) return;
  const int mt = wave / Ntiles;
  const int nt = wave % Ntiles;

  // A 16x32 bf16 layout: lanes 0-15 row=lane, K elems {0..7,16..23}; lanes 16-31 row=lane-16, +8
  const int row = mt * 16 + (lane & 15);
  const int kb  = (lane & 16) ? 8 : 0;
  const unsigned short* __restrict__ arow = A + (size_t)row * K + kb;
  const unsigned short* __restrict__ bptr = Bpk + ((size_t)nt * Ktiles * 32 + lane) * 16;

  v8f acc = {};
  for (int kt = 0; kt < Ktiles; ++kt) {
    const unsigned short* ap = arow + kt * 32;
    const unsigned short* bp = bptr + (size_t)kt * 512;   // 32 lanes * 16 elems per k-tile
    __builtin_prefetch(ap + 64, 0, 1);                    // global_prefetch_b8, two k-tiles ahead
    u16x8 alo = *(const u16x8*)(ap);
    u16x8 ahi = *(const u16x8*)(ap + 16);
    u16x8 blo = *(const u16x8*)(bp);
    u16x8 bhi = *(const u16x8*)(bp + 8);
    u16x16 au, bu;
#pragma unroll
    for (int i = 0; i < 8; ++i) {
      au[i] = alo[i]; au[i + 8] = ahi[i];
      bu[i] = blo[i]; bu[i + 8] = bhi[i];
    }
    v16bf av = __builtin_bit_cast(v16bf, au);
    v16bf bv = __builtin_bit_cast(v16bf, bu);
    acc = __builtin_amdgcn_wmma_f32_16x16x32_bf16(false, av, false, bv,
                                                  (short)0, acc, false, false);
  }

  // D 16x16 f32 layout: lanes 0-15 -> M = vgpr, lanes 16-31 -> M = 8 + vgpr; N = lane&15
  const int rbase = mt * 16 + ((lane & 16) ? 8 : 0);
  const int col   = nt * 16 + (lane & 15);
#pragma unroll
  for (int i = 0; i < 8; ++i)
    C[(size_t)(rbase + i) * Nout + col] = acc[i];
}

// ---------- stage 2: attention scores el/er [N,H] ----------
__global__ void k_attn_scores(const float* __restrict__ feat,
                              const float* __restrict__ al, const float* __restrict__ ar,
                              float* __restrict__ el, float* __restrict__ er, int Nn) {
  int idx = blockIdx.x * blockDim.x + threadIdx.x;
  if (idx >= Nn * HEADS) return;
  int h = idx & 3, n = idx >> 2;
  const float4* f = (const float4*)(feat + (size_t)n * FOUT + h * DHEAD);
  const float4* a = (const float4*)(al + h * DHEAD);
  const float4* b = (const float4*)(ar + h * DHEAD);
  float sl = 0.f, sr = 0.f;
#pragma unroll 8
  for (int i = 0; i < DHEAD / 4; ++i) {
    float4 fv = f[i], av = a[i], bv = b[i];
    sl += fv.x * av.x + fv.y * av.y + fv.z * av.z + fv.w * av.w;
    sr += fv.x * bv.x + fv.y * bv.y + fv.z * bv.z + fv.w * bv.w;
  }
  el[idx] = sl; er[idx] = sr;
}

// ---------- stage 3: edge logits + segment max (atomic u32 on ordered bits) ----------
__global__ void k_edge_logits_max(const int* __restrict__ src, const int* __restrict__ dst,
                                  const float* __restrict__ el, const float* __restrict__ er,
                                  float* __restrict__ ebuf, unsigned int* __restrict__ emax) {
  long long idx = (long long)blockIdx.x * blockDim.x + threadIdx.x;
  if (idx >= (long long)N_EDGES * HEADS) return;
  int h = (int)(idx & 3);
  int e = (int)(idx >> 2);
  float v = el[(size_t)src[e] * HEADS + h] + er[(size_t)dst[e] * HEADS + h];
  v = (v > 0.f) ? v : 0.2f * v;                // leaky_relu(0.2)
  ebuf[idx] = v;
  atomicMax(&emax[(size_t)dst[e] * HEADS + h], ford(v));
}

// ---------- stage 4: exp + segment sum ----------
__global__ void k_edge_exp_sum(const int* __restrict__ dst,
                               const unsigned int* __restrict__ emax,
                               float* __restrict__ ebuf, float* __restrict__ denom) {
  long long idx = (long long)blockIdx.x * blockDim.x + threadIdx.x;
  if (idx >= (long long)N_EDGES * HEADS) return;
  int h = (int)(idx & 3);
  int e = (int)(idx >> 2);
  size_t di = (size_t)dst[e] * HEADS + h;
  float ee = __expf(ebuf[idx] - funord(emax[di]));
  ebuf[idx] = ee;
  atomicAdd(&denom[di], ee);
}

// ---------- stage 5: weighted scatter-aggregate ----------
__global__ void k_edge_aggregate(const int* __restrict__ src, const int* __restrict__ dst,
                                 const float* __restrict__ ebuf, const float* __restrict__ denom,
                                 const float* __restrict__ feat, float* __restrict__ agg) {
  long long idx = (long long)blockIdx.x * blockDim.x + threadIdx.x;
  if (idx >= (long long)N_EDGES * HEADS * (DHEAD / 4)) return;
  int c = (int)(idx & 31);           // 4-float chunk within head dim
  int h = (int)((idx >> 5) & 3);
  int e = (int)(idx >> 7);
  int s = src[e], d = dst[e];
  float a = ebuf[(size_t)e * HEADS + h] / denom[(size_t)d * HEADS + h];
  const float4 fv = *(const float4*)(feat + (size_t)s * FOUT + h * DHEAD + c * 4);
  float* o = agg + (size_t)d * FOUT + h * DHEAD + c * 4;
  atomicAdd(o + 0, a * fv.x);
  atomicAdd(o + 1, a * fv.y);
  atomicAdd(o + 2, a * fv.z);
  atomicAdd(o + 3, a * fv.w);
}

// ---------- stage 6: bias + relu + head-mean; emit f32 and bf16 ----------
__global__ void k_finalize(const float* __restrict__ agg, const float* __restrict__ bias,
                           float* __restrict__ hcur, unsigned short* __restrict__ hbf, int Nn) {
  long long idx = (long long)blockIdx.x * blockDim.x + threadIdx.x;
  if (idx >= (long long)Nn * DHEAD) return;
  int d = (int)(idx & (DHEAD - 1));
  int n = (int)(idx >> 7);
  float s = 0.f;
#pragma unroll
  for (int h = 0; h < HEADS; ++h) {
    float v = agg[(size_t)n * FOUT + h * DHEAD + d] + bias[h * DHEAD + d];
    s += (v > 0.f) ? v : 0.f;
  }
  float hv = s * (1.f / HEADS);
  hcur[idx] = hv;
  hbf[idx]  = f2bf(hv);
}

// ---------- stage 7: node-mean readout (two stage) ----------
__global__ void k_node_sum(const float* __restrict__ hcur, float* __restrict__ racc, int Nn) {
  int d = threadIdx.x;               // 128 threads
  float s = 0.f;
  for (int n = blockIdx.x; n < Nn; n += gridDim.x)
    s += hcur[(size_t)n * DHEAD + d];
  atomicAdd(&racc[d], s);
}

__global__ void k_readout(const float* __restrict__ racc, const float* __restrict__ Wc,
                          const float* __restrict__ bc, float* __restrict__ out) {
  __shared__ float sh[DHEAD];
  int t = threadIdx.x;
  sh[t] = racc[t] * (1.f / (float)N_NODES) * Wc[t];
  __syncthreads();
  for (int s = DHEAD / 2; s > 0; s >>= 1) {
    if (t < s) sh[t] += sh[t + s];
    __syncthreads();
  }
  if (t == 0) out[0] = 1.f / (1.f + __expf(-(sh[0] + bc[0])));
}

// ---------- host driver ----------
extern "C" void kernel_launch(void* const* d_in, const int* in_sizes, int n_in,
                              void* d_out, int out_size, void* d_ws, size_t ws_size,
                              hipStream_t stream) {
  (void)in_sizes; (void)n_in; (void)out_size; (void)ws_size;
  const float* nfeats = (const float*)d_in[0];
  const float* W[3]  = {(const float*)d_in[1], (const float*)d_in[5], (const float*)d_in[9]};
  const float* al[3] = {(const float*)d_in[2], (const float*)d_in[6], (const float*)d_in[10]};
  const float* ar[3] = {(const float*)d_in[3], (const float*)d_in[7], (const float*)d_in[11]};
  const float* bs[3] = {(const float*)d_in[4], (const float*)d_in[8], (const float*)d_in[12]};
  const float* Wc = (const float*)d_in[13];
  const float* bc = (const float*)d_in[14];
  const int* src = (const int*)d_in[15];
  const int* dst = (const int*)d_in[16];
  float* out = (float*)d_out;

  // ---- carve workspace ----
  char* ws = (char*)d_ws;
  size_t off = 0;
  auto carve = [&](size_t bytes) -> void* {
    off = (off + 255) & ~(size_t)255;
    void* p = ws + off;
    off += bytes;
    return p;
  };
  float*          feat  = (float*)carve((size_t)N_NODES * FOUT * 4);   // 102.4 MB
  float*          agg   = (float*)carve((size_t)N_NODES * FOUT * 4);   // 102.4 MB
  unsigned short* xbf   = (unsigned short*)carve((size_t)N_NODES * FIN * 2); // 25.6 MB
  float*          hcur  = (float*)carve((size_t)N_NODES * DHEAD * 4);  // 25.6 MB
  float*          ebuf  = (float*)carve((size_t)N_EDGES * HEADS * 4);  // 12.8 MB
  float*          el    = (float*)carve((size_t)N_NODES * HEADS * 4);
  float*          er    = (float*)carve((size_t)N_NODES * HEADS * 4);
  unsigned int*   emax  = (unsigned int*)carve((size_t)N_NODES * HEADS * 4);
  float*          denom = (float*)carve((size_t)N_NODES * HEADS * 4);
  unsigned short* Bpk   = (unsigned short*)carve((size_t)(FOUT / 16) * (FIN / 32) * 32 * 16 * 2);
  float*          racc  = (float*)carve((size_t)DHEAD * 4);

  const int TB = 256;
  const long long EH  = (long long)N_EDGES * HEADS;
  const long long EHC = (long long)N_EDGES * HEADS * (DHEAD / 4);

  // layer-1 input -> bf16
  {
    long long n = (long long)N_NODES * FIN;
    k_f32_to_bf16<<<(unsigned)((n + TB - 1) / TB), TB, 0, stream>>>(nfeats, xbf, n);
  }

  for (int l = 0; l < 3; ++l) {
    const int K = (l == 0) ? FIN : DHEAD;
    const int Ktiles = K / 32;
    const int Mtiles = N_NODES / 16;       // 3125, exact
    const int Ntiles = FOUT / 16;          // 32

    // pack weights into WMMA B layout
    {
      int total = Ntiles * Ktiles * 32;
      k_pack_b<<<(total + TB - 1) / TB, TB, 0, stream>>>(W[l], Bpk, Ktiles, Ntiles, FOUT);
    }
    // GEMM: feat = x @ W
    {
      int waves = Mtiles * Ntiles;
      int blocks = (waves * 32 + TB - 1) / TB;
      k_gemm_wmma_bf16<<<blocks, TB, 0, stream>>>(xbf, Bpk, feat, Mtiles, Ktiles, Ntiles, K, FOUT);
    }
    // attention scores
    {
      int total = N_NODES * HEADS;
      k_attn_scores<<<(total + TB - 1) / TB, TB, 0, stream>>>(feat, al[l], ar[l], el, er, N_NODES);
    }
    // zero accumulators (graph-capturable memsets)
    hipMemsetAsync(emax,  0, (size_t)N_NODES * HEADS * 4, stream);
    hipMemsetAsync(denom, 0, (size_t)N_NODES * HEADS * 4, stream);
    hipMemsetAsync(agg,   0, (size_t)N_NODES * FOUT  * 4, stream);
    // edge softmax + aggregate
    k_edge_logits_max<<<(unsigned)((EH + TB - 1) / TB), TB, 0, stream>>>(src, dst, el, er, ebuf, emax);
    k_edge_exp_sum  <<<(unsigned)((EH + TB - 1) / TB), TB, 0, stream>>>(dst, emax, ebuf, denom);
    k_edge_aggregate<<<(unsigned)((EHC + TB - 1) / TB), TB, 0, stream>>>(src, dst, ebuf, denom, feat, agg);
    // bias + relu + head mean -> hcur (f32) and xbf (bf16 input for next layer)
    {
      long long total = (long long)N_NODES * DHEAD;
      k_finalize<<<(unsigned)((total + TB - 1) / TB), TB, 0, stream>>>(agg, bs[l], hcur, xbf, N_NODES);
    }
  }

  // readout: mean over nodes, dot with Wc, sigmoid
  hipMemsetAsync(racc, 0, (size_t)DHEAD * 4, stream);
  k_node_sum<<<256, DHEAD, 0, stream>>>(hcur, racc, N_NODES);
  k_readout<<<1, DHEAD, 0, stream>>>(racc, Wc, bc, out);
}
